// GroupQueryAttention_45715631899268
// MI455X (gfx1250) — matrix operand order repaired and checked by reading, not compiled, not run on hardware
//
#include <hip/hip_runtime.h>
#include <hip/hip_bf16.h>
#include <stdint.h>

// Problem constants
#define CB   2
#define CS   2048
#define CDIN 2048
#define CDOUT 2048
#define CNH  16
#define CNKV 4
#define CHD  128
#define CGS  4
#define ATT_SCALE 0.08838834764831845f  // 1/sqrt(128)

typedef __attribute__((ext_vector_type(16))) __bf16 v16bf;
typedef __attribute__((ext_vector_type(8)))  float  v8f;

union Frag {
    v16bf v;
    uint4 q[2];
    unsigned short s[16];
};

// --- async global->LDS (CDNA5) with safe fallback -------------------------
#if defined(__has_builtin)
#if __has_builtin(__builtin_amdgcn_global_load_async_to_lds_b128)
#define HAVE_ASYNC_LDS 1
#endif
#endif

#ifdef HAVE_ASYNC_LDS
typedef __attribute__((ext_vector_type(4))) int v4i_t;
typedef __attribute__((address_space(1))) v4i_t g_v4i;   // global int4
typedef __attribute__((address_space(3))) v4i_t l_v4i;   // LDS int4
__device__ __forceinline__ void async_b128(const void* gsrc, void* ldst) {
    __builtin_amdgcn_global_load_async_to_lds_b128((g_v4i*)gsrc, (l_v4i*)ldst, 0, 0);
}
__device__ __forceinline__ void wait_async0() {
#if __has_builtin(__builtin_amdgcn_s_wait_asynccnt)
    __builtin_amdgcn_s_wait_asynccnt(0);
#else
    asm volatile("s_wait_asynccnt 0x0" ::: "memory");
#endif
}
#endif

__device__ __forceinline__ unsigned short f32_to_bf16(float f) {
    union { float f; uint32_t u; } c; c.f = f;
    uint32_t u = c.u;
    uint32_t r = u + 0x7fffu + ((u >> 16) & 1u);   // round-to-nearest-even
    return (unsigned short)(r >> 16);
}
__device__ __forceinline__ float bf16_to_f32(unsigned short h) {
    union { uint32_t u; float f; } c; c.u = ((uint32_t)h) << 16;
    return c.f;
}

// ---------------------------------------------------------------------------
// f32 -> bf16 elementwise conversion
// ---------------------------------------------------------------------------
__global__ void cvt_bf16_kernel(const float* __restrict__ src,
                                unsigned short* __restrict__ dst, int n) {
    int i = blockIdx.x * blockDim.x + threadIdx.x;
    if (i < n) dst[i] = f32_to_bf16(src[i]);
}

// ---------------------------------------------------------------------------
// bf16 GEMM: C[M,N] = A[M,K] * B[K,N], f32 accumulate via v_wmma_f32_16x16x32_bf16
// 128x128 block tile, 4 waves (2x2), 64x64 per wave, K-step 32.
// C_F32 templated so the epilogue is branch-free.
// ---------------------------------------------------------------------------
template <bool C_F32>
__global__ __launch_bounds__(128)
void gemm_bf16_kernel(const unsigned short* __restrict__ A,
                      const unsigned short* __restrict__ B,
                      void* __restrict__ Cout,
                      int M, int N, int K) {
    __shared__ __align__(16) unsigned short As[128][32];  // row-major A tile
    __shared__ __align__(16) unsigned short Bt[128][32];  // transposed B tile: [n][k]

    const int tid  = threadIdx.x;
    const int lane = tid & 31;
    const int wave = tid >> 5;            // 0..3
    const int l16  = lane & 15;
    const int half = lane >> 4;
    const int wm   = (wave >> 1) * 64;
    const int wn   = (wave & 1) * 64;
    const int bm   = blockIdx.y * 128;
    const int bn   = blockIdx.x * 128;

    v8f acc[4][4];
#pragma unroll
    for (int i = 0; i < 4; ++i)
#pragma unroll
        for (int j = 0; j < 4; ++j)
            acc[i][j] = (v8f){0.f,0.f,0.f,0.f,0.f,0.f,0.f,0.f};

    for (int k0 = 0; k0 < K; k0 += 32) {
        __syncthreads();
        // A tile: 128x32 bf16, 512 16B chunks, 4 per thread (coalesced rows)
#pragma unroll
        for (int i = 0; i < 4; ++i) {
            int c   = tid + i * 128;
            int row = c >> 2;
            int col = (c & 3) * 8;
            const void* src = &A[(size_t)(bm + row) * K + k0 + col];
#ifdef HAVE_ASYNC_LDS
            async_b128(src, &As[row][col]);
#else
            *(uint4*)&As[row][col] = *(const uint4*)src;
#endif
        }
        // B tile transposed: thread t owns column n = bn+t; k-loop reads are
        // contiguous 256B across the workgroup per iteration (coalesced).
#pragma unroll
        for (int k = 0; k < 32; ++k)
            Bt[tid][k] = B[(size_t)(k0 + k) * N + bn + tid];
#ifdef HAVE_ASYNC_LDS
        wait_async0();
#endif
        __syncthreads();

        Frag af[4], bf[4];
#pragma unroll
        for (int t = 0; t < 4; ++t) {
            int row = wm + t * 16 + l16;
            af[t].q[0] = *(const uint4*)&As[row][half * 8];
            af[t].q[1] = *(const uint4*)&As[row][16 + half * 8];
        }
#pragma unroll
        for (int t = 0; t < 4; ++t) {
            int col = wn + t * 16 + l16;
            bf[t].q[0] = *(const uint4*)&Bt[col][half * 8];
            bf[t].q[1] = *(const uint4*)&Bt[col][16 + half * 8];
        }
#pragma unroll
        for (int i = 0; i < 4; ++i)
#pragma unroll
            for (int j = 0; j < 4; ++j)
                acc[i][j] = __builtin_amdgcn_wmma_f32_16x16x32_bf16(
                    false, af[i].v, false, bf[j].v, (short)0, acc[i][j],
                    false, false);
    }

    // Epilogue: C/D layout -> lane (n=l16, half) holds rows r + half*8.
    if (C_F32) {
        float* cp = (float*)Cout + (size_t)(bm + wm + half * 8) * N + (bn + wn + l16);
#pragma unroll
        for (int i = 0; i < 4; ++i)
#pragma unroll
            for (int r = 0; r < 8; ++r) {
                float* rowp = cp + (size_t)(i * 16 + r) * N;
#pragma unroll
                for (int j = 0; j < 4; ++j)
                    rowp[j * 16] = acc[i][j][r];
            }
    } else {
        unsigned short* cp = (unsigned short*)Cout +
                             (size_t)(bm + wm + half * 8) * N + (bn + wn + l16);
#pragma unroll
        for (int i = 0; i < 4; ++i)
#pragma unroll
            for (int r = 0; r < 8; ++r) {
                unsigned short* rowp = cp + (size_t)(i * 16 + r) * N;
#pragma unroll
                for (int j = 0; j < 4; ++j)
                    rowp[j * 16] = f32_to_bf16(acc[i][j][r]);
            }
    }
}

// ---------------------------------------------------------------------------
// RoPE + relayout: in [B*S, H*HD] bf16 -> out [B, H, S, HD] bf16
// ---------------------------------------------------------------------------
__global__ void rope_kernel(const unsigned short* __restrict__ in,
                            const float* __restrict__ cosT,
                            const float* __restrict__ sinT,
                            unsigned short* __restrict__ out, int H) {
    int idx = blockIdx.x * blockDim.x + threadIdx.x;
    int total = CB * CS * H * 64;
    if (idx >= total) return;
    int d  = idx & 63;
    int i2 = idx >> 6;
    int h  = i2 % H;
    int s  = (i2 / H) % CS;
    int b  = i2 / (H * CS);

    const unsigned short* row = in + ((size_t)(b * CS + s) * H + h) * CHD;
    float t1 = bf16_to_f32(row[d]);
    float t2 = bf16_to_f32(row[d + 64]);
    float c1 = cosT[s * CHD + d],      s1 = sinT[s * CHD + d];
    float c2 = cosT[s * CHD + d + 64], s2 = sinT[s * CHD + d + 64];
    float o1 = t1 * c1 - t2 * s1;
    float o2 = t2 * c2 + t1 * s2;

    size_t ob = (((size_t)b * H + h) * CS + s) * CHD;
    out[ob + d]      = f32_to_bf16(o1);
    out[ob + d + 64] = f32_to_bf16(o2);
}

// ---------------------------------------------------------------------------
// V relayout: [B*S, NKV*HD] bf16 -> [B, NKV, S, HD] bf16
// ---------------------------------------------------------------------------
__global__ void vtrans_kernel(const unsigned short* __restrict__ in,
                              unsigned short* __restrict__ out) {
    int idx = blockIdx.x * blockDim.x + threadIdx.x;
    int total = CB * CS * CNKV * CHD;
    if (idx >= total) return;
    int d  = idx & (CHD - 1);
    int i2 = idx >> 7;
    int h  = i2 % CNKV;
    int s  = (i2 / CNKV) % CS;
    int b  = i2 / (CNKV * CS);
    out[(((size_t)b * CNKV + h) * CS + s) * CHD + d] =
        in[((size_t)(b * CS + s) * CNKV + h) * CHD + d];
}

// ---------------------------------------------------------------------------
// Flash attention (causal, GQA): Q [B,NH,S,HD], K/V [B,NKV,S,HD] all bf16,
// writes ctx [B, S, NH*HD] bf16.  4 waves/WG, 16 q-rows per wave.
// ---------------------------------------------------------------------------
__global__ __launch_bounds__(128)
void attn_kernel(const unsigned short* __restrict__ Q,
                 const unsigned short* __restrict__ Kg,
                 const unsigned short* __restrict__ Vg,
                 unsigned short* __restrict__ ctx) {
    __shared__ __align__(16) unsigned short Ks[32][128];   // [key][hd]
    __shared__ __align__(16) unsigned short Vs[128][32];   // transposed [hd][key]
    __shared__ __align__(16) unsigned short Ps[4][16][32]; // per-wave P staging

    const int tid  = threadIdx.x;
    const int lane = tid & 31;
    const int wave = tid >> 5;
    const int l16  = lane & 15;
    const int half = lane >> 4;

    const int bh  = blockIdx.y;
    const int b   = bh >> 4;
    const int h   = bh & 15;
    const int kvh = h >> 2;                  // GQA: GS=4
    const int q0  = blockIdx.x * 64 + wave * 16;

    const unsigned short* Qb = Q  + ((size_t)(b * CNH  + h)   * CS) * CHD;
    const unsigned short* Kb = Kg + ((size_t)(b * CNKV + kvh) * CS) * CHD;
    const unsigned short* Vb = Vg + ((size_t)(b * CNKV + kvh) * CS) * CHD;

    // Q fragments for HD=128 contraction: 4 x (16x32 bf16 A-fragments)
    Frag qf[4];
#pragma unroll
    for (int j = 0; j < 4; ++j) {
        const unsigned short* qp = Qb + (size_t)(q0 + l16) * CHD + j * 32;
        qf[j].q[0] = *(const uint4*)(qp + half * 8);
        qf[j].q[1] = *(const uint4*)(qp + 16 + half * 8);
    }

    float rmax[8], rsum[8];
    v8f   oacc[8];
#pragma unroll
    for (int r = 0; r < 8; ++r) { rmax[r] = -3.0e38f; rsum[r] = 0.f; }
#pragma unroll
    for (int c = 0; c < 8; ++c)
        oacc[c] = (v8f){0.f,0.f,0.f,0.f,0.f,0.f,0.f,0.f};

    const int kend = blockIdx.x * 64 + 64;   // exclusive causal bound for block
    for (int kt = 0; kt < kend; kt += 32) {
        __syncthreads();
        // K tile 32x128: 512 16B chunks, 4/thread
#pragma unroll
        for (int i = 0; i < 4; ++i) {
            int c   = tid + i * 128;
            int row = c >> 4;
            int col = (c & 15) * 8;
            const void* src = &Kb[(size_t)(kt + row) * CHD + col];
#ifdef HAVE_ASYNC_LDS
            async_b128(src, &Ks[row][col]);
#else
            *(uint4*)&Ks[row][col] = *(const uint4*)src;
#endif
        }
        // V tile transposed: thread owns hd col = tid; coalesced per-k reads
#pragma unroll
        for (int k = 0; k < 32; ++k)
            Vs[tid][k] = Vb[(size_t)(kt + k) * CHD + tid];
#ifdef HAVE_ASYNC_LDS
        wait_async0();
#endif
        __syncthreads();

        if (kt <= q0 + 15) {
            // ---- scores S = Q K^T over 32 keys (two 16x16 tiles) ----
            v8f s0 = (v8f){0.f,0.f,0.f,0.f,0.f,0.f,0.f,0.f};
            v8f s1 = (v8f){0.f,0.f,0.f,0.f,0.f,0.f,0.f,0.f};
#pragma unroll
            for (int j = 0; j < 4; ++j) {
                Frag k0f, k1f;
                k0f.q[0] = *(const uint4*)&Ks[l16][j * 32 + half * 8];
                k0f.q[1] = *(const uint4*)&Ks[l16][j * 32 + 16 + half * 8];
                k1f.q[0] = *(const uint4*)&Ks[16 + l16][j * 32 + half * 8];
                k1f.q[1] = *(const uint4*)&Ks[16 + l16][j * 32 + 16 + half * 8];
                s0 = __builtin_amdgcn_wmma_f32_16x16x32_bf16(
                        false, qf[j].v, false, k0f.v, (short)0, s0, false, false);
                s1 = __builtin_amdgcn_wmma_f32_16x16x32_bf16(
                        false, qf[j].v, false, k1f.v, (short)0, s1, false, false);
            }
            // ---- scale + causal mask + online softmax ----
            float tmax[8];
#pragma unroll
            for (int r = 0; r < 8; ++r) {
                int qrow = q0 + half * 8 + r;
                float v0 = (kt + l16      <= qrow) ? s0[r] * ATT_SCALE : -3.0e38f;
                float v1 = (kt + 16 + l16 <= qrow) ? s1[r] * ATT_SCALE : -3.0e38f;
                s0[r] = v0; s1[r] = v1;
                tmax[r] = fmaxf(v0, v1);
            }
#pragma unroll
            for (int m = 1; m < 16; m <<= 1)
#pragma unroll
                for (int r = 0; r < 8; ++r)
                    tmax[r] = fmaxf(tmax[r], __shfl_xor(tmax[r], m, 32));

            float fac[8], part[8];
#pragma unroll
            for (int r = 0; r < 8; ++r) {
                float nm = fmaxf(rmax[r], tmax[r]);
                fac[r]  = __expf(rmax[r] - nm);
                rmax[r] = nm;
                float p0 = __expf(s0[r] - nm);
                float p1 = __expf(s1[r] - nm);
                s0[r] = p0; s1[r] = p1;
                part[r] = p0 + p1;
            }
#pragma unroll
            for (int m = 1; m < 16; m <<= 1)
#pragma unroll
                for (int r = 0; r < 8; ++r)
                    part[r] += __shfl_xor(part[r], m, 32);
#pragma unroll
            for (int r = 0; r < 8; ++r)
                rsum[r] = rsum[r] * fac[r] + part[r];
#pragma unroll
            for (int c = 0; c < 8; ++c)
#pragma unroll
                for (int r = 0; r < 8; ++r)
                    oacc[c][r] *= fac[r];

            // ---- C-layout P -> A-layout via per-wave LDS staging ----
#pragma unroll
            for (int r = 0; r < 8; ++r) {
                Ps[wave][half * 8 + r][l16]      = f32_to_bf16(s0[r]);
                Ps[wave][half * 8 + r][16 + l16] = f32_to_bf16(s1[r]);
            }
            asm volatile("s_wait_dscnt 0" ::: "memory");  // wave-local LDS RAW
            Frag pf;
            {
                const unsigned short* pp = &Ps[wave][l16][0];
                pf.q[0] = *(const uint4*)(pp + half * 8);
                pf.q[1] = *(const uint4*)(pp + 16 + half * 8);
            }
            // ---- O += P * V over HD (8 n-tiles of 16) ----
#pragma unroll
            for (int c = 0; c < 8; ++c) {
                Frag vf;
                vf.q[0] = *(const uint4*)&Vs[c * 16 + l16][half * 8];
                vf.q[1] = *(const uint4*)&Vs[c * 16 + l16][16 + half * 8];
                oacc[c] = __builtin_amdgcn_wmma_f32_16x16x32_bf16(
                    false, pf.v, false, vf.v, (short)0, oacc[c], false, false);
            }
        }
    }

    // ---- normalize + store ctx [B, S, NH*HD] ----
#pragma unroll
    for (int r = 0; r < 8; ++r) rsum[r] = 1.0f / rsum[r];
    unsigned short* cp = ctx + ((size_t)b * CS + q0 + half * 8) * (size_t)CDOUT
                       + (size_t)h * CHD + l16;
#pragma unroll
    for (int r = 0; r < 8; ++r) {
        unsigned short* rowp = cp + (size_t)r * CDOUT;
#pragma unroll
        for (int c = 0; c < 8; ++c)
            rowp[c * 16] = f32_to_bf16(oacc[c][r] * rsum[r]);
    }
}

// ---------------------------------------------------------------------------
// Host-side orchestration
// ---------------------------------------------------------------------------
extern "C" void kernel_launch(void* const* d_in, const int* in_sizes, int n_in,
                              void* d_out, int out_size, void* d_ws, size_t ws_size,
                              hipStream_t stream) {
    (void)in_sizes; (void)n_in; (void)out_size; (void)ws_size;
    const float* x   = (const float*)d_in[0];
    const float* Wq  = (const float*)d_in[1];
    const float* Wk  = (const float*)d_in[2];
    const float* Wv  = (const float*)d_in[3];
    const float* Wo  = (const float*)d_in[4];
    const float* cosT = (const float*)d_in[5];
    const float* sinT = (const float*)d_in[6];
    float* out = (float*)d_out;

    const size_t M = (size_t)CB * CS;          // 4096 rows
    char* ws = (char*)d_ws;
    size_t off = 0;
    auto take = [&](size_t bytes) -> char* {
        char* p = ws + off;
        off = (off + bytes + 255) & ~(size_t)255;
        return p;
    };
    unsigned short* xbf  = (unsigned short*)take(M * CDIN * 2);
    unsigned short* wqb  = (unsigned short*)take((size_t)CDIN * CDOUT * 2);
    unsigned short* wkb  = (unsigned short*)take((size_t)CDIN * CNKV * CHD * 2);
    unsigned short* wvb  = (unsigned short*)take((size_t)CDIN * CNKV * CHD * 2);
    unsigned short* wob  = (unsigned short*)take((size_t)CDOUT * CDOUT * 2);
    unsigned short* qtmp = (unsigned short*)take(M * CDOUT * 2);
    unsigned short* ktmp = (unsigned short*)take(M * CNKV * CHD * 2);
    unsigned short* vtmp = (unsigned short*)take(M * CNKV * CHD * 2);
    unsigned short* qr   = (unsigned short*)take(M * CDOUT * 2);
    unsigned short* kr   = (unsigned short*)take(M * CNKV * CHD * 2);
    unsigned short* vr   = (unsigned short*)take(M * CNKV * CHD * 2);
    unsigned short* ctxb = (unsigned short*)take(M * CDOUT * 2);

    auto cvt = [&](const float* s, unsigned short* d, int n) {
        cvt_bf16_kernel<<<(n + 255) / 256, 256, 0, stream>>>(s, d, n);
    };
    cvt(x,  xbf, (int)(M * CDIN));
    cvt(Wq, wqb, CDIN * CDOUT);
    cvt(Wk, wkb, CDIN * CNKV * CHD);
    cvt(Wv, wvb, CDIN * CNKV * CHD);
    cvt(Wo, wob, CDOUT * CDOUT);

    // QKV projections (bf16 WMMA, bf16 outputs)
    gemm_bf16_kernel<false><<<dim3(CDOUT / 128, (int)(M / 128)), 128, 0, stream>>>(
        xbf, wqb, qtmp, (int)M, CDOUT, CDIN);
    gemm_bf16_kernel<false><<<dim3((CNKV * CHD) / 128, (int)(M / 128)), 128, 0, stream>>>(
        xbf, wkb, ktmp, (int)M, CNKV * CHD, CDIN);
    gemm_bf16_kernel<false><<<dim3((CNKV * CHD) / 128, (int)(M / 128)), 128, 0, stream>>>(
        xbf, wvb, vtmp, (int)M, CNKV * CHD, CDIN);

    // RoPE + relayout
    {
        int nq = CB * CS * CNH * 64;
        rope_kernel<<<(nq + 255) / 256, 256, 0, stream>>>(qtmp, cosT, sinT, qr, CNH);
        int nk = CB * CS * CNKV * 64;
        rope_kernel<<<(nk + 255) / 256, 256, 0, stream>>>(ktmp, cosT, sinT, kr, CNKV);
        int nv = CB * CS * CNKV * CHD;
        vtrans_kernel<<<(nv + 255) / 256, 256, 0, stream>>>(vtmp, vr);
    }

    // Causal GQA flash attention
    attn_kernel<<<dim3(CS / 64, CB * CNH), 128, 0, stream>>>(qr, kr, vr, ctxb);

    // Output projection, f32 result straight to d_out
    gemm_bf16_kernel<true><<<dim3(CDOUT / 128, (int)(M / 128)), 128, 0, stream>>>(
        ctxb, wob, out, (int)M, CDOUT, CDOUT);
}
